// SelfAttentionDistil_76287209111916
// MI455X (gfx1250) — compile-verified, matrix operand order checked by reading, not run on hardware
//
#include <hip/hip_runtime.h>
#include <math.h>

// ---------------------------------------------------------------------------
// MI455X / gfx1250, wave32. Conv-as-GEMM with v_wmma_f32_16x16x32_f16.
// Panel-resident LDS (256KB of the WGP's 320KB) + async global->LDS staging.
// ---------------------------------------------------------------------------

typedef _Float16 half4_t  __attribute__((ext_vector_type(4)));
typedef _Float16 half8_t  __attribute__((ext_vector_type(8)));
typedef _Float16 half16_t __attribute__((ext_vector_type(16)));
typedef float    float8_t __attribute__((ext_vector_type(8)));

#define B_DIM 16
#define L_DIM 4096
#define C_DIM 512
#define LQ    2048            // L/2, per-stream length
#define TM    128             // tile rows (t)
#define TN    128             // tile cols (co)
#define BN_EPS 1e-5f

#define PANEL_HALFS (128 * 512)          // one panel: 128 rows x 512 halfs
#define PANEL_BYTES (PANEL_HALFS * 2)    // 128 KB

// ---------------------------------------------------------------------------
// CDNA5 async global->LDS copy (ASYNCcnt-tracked), 16B per lane.
// ---------------------------------------------------------------------------
__device__ __forceinline__ void async_load_b128(unsigned lds_off, const void* gaddr) {
  asm volatile("global_load_async_to_lds_b128 %0, %1, off"
               :: "v"(lds_off), "v"(gaddr)
               : "memory");
}
__device__ __forceinline__ void wait_asynccnt0() {
  asm volatile("s_wait_asynccnt 0x0" ::: "memory");
}

// ---------------------------------------------------------------------------
// Kernel 1: weights f32 (co, ci, 3) -> f16 WT[kk][co][ci]
// ---------------------------------------------------------------------------
__global__ void prep_w_kernel(const float* __restrict__ conv_w,
                              _Float16* __restrict__ wt) {
  int idx = blockIdx.x * 256 + threadIdx.x;     // 3*512*512 = 786432
  int ci = idx & 511;
  int co = (idx >> 9) & 511;
  int kk = idx >> 18;
  wt[idx] = (_Float16)conv_w[(co * C_DIM + ci) * 3 + kk];
}

// ---------------------------------------------------------------------------
// Kernel 2: x f32 (B,L,C) -> f16 XH[b][s][t][ci] (de-interleave streams)
// ---------------------------------------------------------------------------
__global__ void prep_x_kernel(const float* __restrict__ x,
                              _Float16* __restrict__ xh) {
  long idx4 = ((long)(blockIdx.x * 256 + threadIdx.x)) * 4; // groups of 4 chans
  int  c  = (int)(idx4 & 511);
  long r  = idx4 >> 9;          // (b*2+s)*2048 + t
  int  t  = (int)(r & 2047);
  int  bs = (int)(r >> 11);
  int  s  = bs & 1;
  int  b  = bs >> 1;
  long l  = 2L * t + s;
  const float4 v = *(const float4*)(x + ((long)b * L_DIM + l) * C_DIM + c);
  half4_t h;
  h[0] = (_Float16)v.x; h[1] = (_Float16)v.y;
  h[2] = (_Float16)v.z; h[3] = (_Float16)v.w;
  *(half4_t*)(xh + idx4) = h;
}

// ---------------------------------------------------------------------------
// LDS fragment load: panel stored [row][512] halfs, XOR-16 swizzle on the
// 8-half (16B) chunk index (row stride 1KB == 0 mod banks, so swizzle must
// permute across 16 chunks to keep frag loads conflict-free).
// Fragment (16x32 f16, ISA layout):
//   lanes 0-15 : row = m0+lane,    chunks {c0, c0+2}  (K{0..7},K{16..23})
//   lanes 16-31: row = m0+lane-16, chunks {c0, c0+2}  with c0 = 4*kc+1
// ---------------------------------------------------------------------------
__device__ __forceinline__ half16_t load_frag(const _Float16* base, int row, int c0) {
  int swz = row & 15;
  half8_t lo = *(const half8_t*)(base + row * C_DIM + (((c0    ) ^ swz) * 8));
  half8_t hh = *(const half8_t*)(base + row * C_DIM + (((c0 + 2) ^ swz) * 8));
  half16_t f;
#pragma unroll
  for (int e = 0; e < 8; ++e) { f[e] = lo[e]; f[e + 8] = hh[e]; }
  return f;
}

// ---------------------------------------------------------------------------
// Kernel 3: per (b,s) GEMM  D[t,co] = sum_kk sum_ci XH[(t+kk-1)%2048][ci] *
// WT[kk][co][ci], fused bias + BN + ELU + residual -> xr[b][s][t][co] f32.
// Block: 256 thr (8 waves), tile 128(t) x 128(co); wave = 64x32 = 4x2 wmma.
// Per tap kk: stage full K=512 A/B panels (256KB LDS) with async-to-LDS DMA,
// then 16 barrier-free K-steps of pure ds_load_b128 + v_wmma.
// ---------------------------------------------------------------------------
__global__ __launch_bounds__(256)
void conv_gemm_kernel(const _Float16* __restrict__ xh,
                      const _Float16* __restrict__ wt,
                      const float* __restrict__ x,
                      const float* __restrict__ conv_b,
                      const float* __restrict__ gamma,
                      const float* __restrict__ beta,
                      const float* __restrict__ rmean,
                      const float* __restrict__ rvar,
                      float* __restrict__ xr) {
  extern __shared__ _Float16 smem[];              // 256KB dynamic LDS
  _Float16* As = smem;                            // A panel: t rows
  _Float16* Bs = smem + PANEL_HALFS;              // B panel: co rows
  const unsigned ldsA = (unsigned)(uintptr_t)As;  // LDS byte offset (low 32b)
  const unsigned ldsB = (unsigned)(uintptr_t)Bs;

  const int tid  = threadIdx.x;
  const int lane = tid & 31;
  const int wave = tid >> 5;      // 0..7
  const int wm   = wave >> 2;     // 0..1 : 64-row slab of t
  const int wn   = wave & 3;      // 0..3 : 32-col slab of co

  const int bs = blockIdx.z;      // b*2+s
  const int s  = bs & 1;
  const int b  = bs >> 1;
  const int t0 = blockIdx.x * TM;
  const int n0 = blockIdx.y * TN;

  const _Float16* xbs = xh + (size_t)bs * (LQ * C_DIM);

  float8_t acc[4][2];
#pragma unroll
  for (int i = 0; i < 4; ++i)
#pragma unroll
    for (int j = 0; j < 2; ++j)
#pragma unroll
      for (int e = 0; e < 8; ++e) acc[i][j][e] = 0.0f;

  const int mloc = lane & 15;
  const int hi   = lane >> 4;     // 0: K{0..7,16..23}; 1: K{8..15,24..31}

  for (int kk = 0; kk < 3; ++kk) {
    const _Float16* wk = wt + (size_t)kk * (C_DIM * C_DIM);

    // ---- stage both 128x512 panels via async global->LDS DMA -------------
    // 8192 16B-chunks per panel, 256 threads -> 32 chunks each per panel.
#pragma unroll 4
    for (int i = 0; i < 32; ++i) {
      int g   = i * 256 + tid;            // 0..8191
      int row = g >> 6;                   // 0..127
      int c   = g & 63;                   // 16B chunk within row
      unsigned loff = (unsigned)(row * 1024 + ((c ^ (row & 15)) * 16));
      int tg = (t0 + row + kk - 1) & (LQ - 1);     // circular shift on t
      async_load_b128(ldsA + loff, xbs + (size_t)tg * C_DIM + c * 8);
      async_load_b128(ldsB + loff, wk + (size_t)(n0 + row) * C_DIM + c * 8);
    }
    wait_asynccnt0();
    __syncthreads();

    // ---- 16 barrier-free K-steps: ds_load_b128 + v_wmma only -------------
    for (int kc = 0; kc < C_DIM / 32; ++kc) {
      const int c0 = kc * 4 + hi;
      half16_t afr[4], bfr[2];
#pragma unroll
      for (int i = 0; i < 4; ++i)
        afr[i] = load_frag(As, wm * 64 + i * 16 + mloc, c0);
#pragma unroll
      for (int j = 0; j < 2; ++j)
        bfr[j] = load_frag(Bs, wn * 32 + j * 16 + mloc, c0);

#pragma unroll
      for (int i = 0; i < 4; ++i)
#pragma unroll
        for (int j = 0; j < 2; ++j)
          acc[i][j] = __builtin_amdgcn_wmma_f32_16x16x32_f16(
              false, afr[i], false, bfr[j], (short)0, acc[i][j], false, false);
    }
    __syncthreads();                      // all reads done before re-staging
  }

  // -------- fused epilogue: bias + BN + ELU + residual --------
#pragma unroll
  for (int j = 0; j < 2; ++j) {
    const int co = n0 + wn * 32 + j * 16 + (lane & 15);
    const float cb    = conv_b[co];
    const float mu    = rmean[co];
    const float scale = __frsqrt_rn(rvar[co] + BN_EPS) * gamma[co];
    const float bt    = beta[co];
#pragma unroll
    for (int i = 0; i < 4; ++i) {
#pragma unroll
      for (int r = 0; r < 8; ++r) {
        // C/D layout: lanes 0-15 hold M = r, lanes 16-31 hold M = r+8
        int tglob = t0 + wm * 64 + i * 16 + r + ((lane >> 4) << 3);
        float v = acc[i][j][r] + cb;
        v = (v - mu) * scale + bt;
        v = v > 0.0f ? v : (__expf(v) - 1.0f);             // ELU
        int l = 2 * tglob + s;
        v += x[((size_t)b * L_DIM + l) * C_DIM + co];      // residual
        xr[((size_t)bs * LQ + tglob) * C_DIM + co] = v;
      }
    }
  }
}

// ---------------------------------------------------------------------------
// Kernel 4: maxpool(k=3, s=2, pad=1) along L on interleaved streams:
//   out[b][j][co] = max(xr[b][1][j-1][co] (or -inf), xr[b][0][j][co],
//                       xr[b][1][j][co]);  xr already [.., t, co].
// ---------------------------------------------------------------------------
__global__ void pool_kernel(const float* __restrict__ xr,
                            float* __restrict__ out) {
  long idx4 = ((long)(blockIdx.x * 256 + threadIdx.x)) * 4;
  int  c = (int)(idx4 & 511);
  long r = idx4 >> 9;             // b*2048 + j
  int  j = (int)(r & 2047);
  int  b = (int)(r >> 11);
  const float* x0 = xr + ((size_t)(b * 2 + 0) * LQ + j) * C_DIM + c;
  const float* x1 = xr + ((size_t)(b * 2 + 1) * LQ + j) * C_DIM + c;
  float4 v0 = *(const float4*)x0;
  float4 v1 = *(const float4*)x1;
  float4 vp;
  if (j > 0) {
    vp = *(const float4*)(x1 - C_DIM);
  } else {
    vp.x = -__builtin_inff(); vp.y = -__builtin_inff();
    vp.z = -__builtin_inff(); vp.w = -__builtin_inff();
  }
  float4 o;
  o.x = fmaxf(fmaxf(v0.x, v1.x), vp.x);
  o.y = fmaxf(fmaxf(v0.y, v1.y), vp.y);
  o.z = fmaxf(fmaxf(v0.z, v1.z), vp.z);
  o.w = fmaxf(fmaxf(v0.w, v1.w), vp.w);
  *(float4*)(out + idx4) = o;
}

// ---------------------------------------------------------------------------
// Launch. Workspace layout (bytes):
//   [0)              WT  f16 : 3*512*512*2      = 1.5 MB
//   [WT end)         XH  f16 : 16*2*2048*512*2  = 64 MB
//   [XH end)         XR  f32 : 16*2*2048*512*4  = 128 MB   (~203 MB total)
// ---------------------------------------------------------------------------
extern "C" void kernel_launch(void* const* d_in, const int* in_sizes, int n_in,
                              void* d_out, int out_size, void* d_ws, size_t ws_size,
                              hipStream_t stream) {
  const float* x      = (const float*)d_in[0];
  const float* conv_w = (const float*)d_in[1];
  const float* conv_b = (const float*)d_in[2];
  // d_in[3] = pad_w, d_in[4] = pad_b : unused by the reference
  const float* gamma  = (const float*)d_in[5];
  const float* beta   = (const float*)d_in[6];
  const float* rmean  = (const float*)d_in[7];
  const float* rvar   = (const float*)d_in[8];
  float*       out    = (float*)d_out;

  char* ws = (char*)d_ws;
  const size_t wt_bytes = (size_t)3 * C_DIM * C_DIM * 2;
  const size_t xh_bytes = (size_t)B_DIM * 2 * LQ * C_DIM * 2;
  _Float16* wt = (_Float16*)ws;
  _Float16* xh = (_Float16*)(ws + wt_bytes);
  float*    xr = (float*)(ws + wt_bytes + xh_bytes);

  prep_w_kernel<<<(3 * C_DIM * C_DIM) / 256, 256, 0, stream>>>(conv_w, wt);
  prep_x_kernel<<<((long)B_DIM * 2 * LQ * C_DIM) / (256 * 4), 256, 0, stream>>>(x, xh);

  dim3 grid(LQ / TM, C_DIM / TN, B_DIM * 2);   // (16, 4, 32)
  conv_gemm_kernel<<<grid, 256, 2 * PANEL_BYTES, stream>>>(
      xh, wt, x, conv_b, gamma, beta, rmean, rvar, xr);

  pool_kernel<<<((long)B_DIM * LQ * C_DIM) / (256 * 4), 256, 0, stream>>>(xr, out);
}
